// ProteinBranchGNN_23072564314613
// MI455X (gfx1250) — matrix operand order repaired and checked by reading, not compiled, not run on hardware
//
#include <hip/hip_runtime.h>
#include <hip/hip_bf16.h>

// ---- CDNA5 WMMA vector types --------------------------------------------
typedef __attribute__((ext_vector_type(16))) __bf16 v16bf;
typedef __attribute__((ext_vector_type(8)))  __bf16 v8bf;
typedef __attribute__((ext_vector_type(8)))  float  v8f;
typedef __attribute__((ext_vector_type(4)))  unsigned int u32x4;
typedef __attribute__((ext_vector_type(8)))  int i32x8;
typedef __attribute__((ext_vector_type(4)))  int i32x4;

#define B_GR     16
#define N_NODES  4096
#define E_EDGES  65536
#define TOTAL    (B_GR * N_NODES)     // 65536 nodes
#define CH       128
#define BN_EPS   1e-5f

// Detect the TDM builtin (tensor_load_to_lds). Arity differs per toolchain:
// ROCm 7.2 clang-22 -> 5 args; therock/clang-23 (ships the TDM header) -> 6.
#if defined(__has_builtin)
#if __has_builtin(__builtin_amdgcn_tensor_load_to_lds) && __has_builtin(__builtin_amdgcn_s_wait_tensorcnt)
#define GNN_USE_TDM 1
#endif
#endif
#ifndef GNN_USE_TDM
#define GNN_USE_TDM 0
#endif

// =========================================================================
// Pack a row-major f32 weight [K, 128] into the per-lane bf16 B-fragment
// layout for v_wmma_f32_16x16x32_bf16 (wave32).
// B fragment (16-bit, 32x16 tile): lanes 0-15 hold K=0..7,16..23 of column n,
// lanes 16-31 hold K=8..15,24..31.
// Linear dst index = ((ntile*ksteps + kstep)*32 + lane)*16 + i.
// =========================================================================
__global__ void gnn_pack_b_kernel(const float* __restrict__ src,
                                  __bf16* __restrict__ dst, int K) {
    int ksteps = K >> 5;
    int total  = 8 * ksteps * 32 * 16;
    int idx = blockIdx.x * blockDim.x + threadIdx.x;
    if (idx >= total) return;
    int i    = idx & 15;
    int lane = (idx >> 4) & 31;
    int rest = idx >> 9;               // ntile*ksteps + kstep
    int ks   = rest % ksteps;
    int nt   = rest / ksteps;
    int n    = nt * 16 + (lane & 15);
    int k    = ks * 32 + ((lane < 16) ? 0 : 8) + ((i < 8) ? i : i + 8);
    dst[idx] = (__bf16)src[(size_t)k * CH + n];
}

// =========================================================================
// Fused embed: t = relu(x*w1 + b1)  [65536 x 64],  h = t @ w2 + b2.
// One wave -> one 16x16 tile of h via 2x v_wmma_f32_16x16x32_bf16.
// =========================================================================
__global__ __launch_bounds__(256)
void gnn_embed_kernel(const float* __restrict__ x,
                      const float* __restrict__ w1, const float* __restrict__ b1,
                      const __bf16* __restrict__ w2pk, const float* __restrict__ b2,
                      float* __restrict__ hf, __bf16* __restrict__ hb) {
    int lane  = threadIdx.x & 31;
    int ntile = threadIdx.x >> 5;      // 0..7  (8 waves/block)
    int mtile = blockIdx.x;            // 0..4095
    int m     = mtile * 16 + (lane & 15);
    float xm  = x[m];
    v8f acc = {};
#pragma unroll
    for (int ks = 0; ks < 2; ++ks) {   // K = 64
        int kb0 = ks * 32 + ((lane < 16) ? 0 : 8);
        int kb1 = kb0 + 16;
        v16bf a;
#pragma unroll
        for (int i = 0; i < 8; ++i) {
            float t0 = fmaxf(fmaf(xm, w1[kb0 + i], b1[kb0 + i]), 0.0f);
            float t1 = fmaxf(fmaf(xm, w1[kb1 + i], b1[kb1 + i]), 0.0f);
            a[i]     = (__bf16)t0;
            a[i + 8] = (__bf16)t1;
        }
        v16bf b = *(const v16bf*)(w2pk + (size_t)((ntile * 2 + ks) * 32 + lane) * 16);
        acc = __builtin_amdgcn_wmma_f32_16x16x32_bf16(false, a, false, b,
                                                      (short)0, acc, false, false);
    }
    int n     = ntile * 16 + (lane & 15);
    int mbase = mtile * 16 + ((lane < 16) ? 0 : 8);
#pragma unroll
    for (int j = 0; j < 8; ++j) {
        float v = acc[j] + b2[n];
        size_t o = (size_t)(mbase + j) * CH + n;
        hf[o] = v;
        hb[o] = (__bf16)v;
    }
}

// =========================================================================
// GEMM (K = 128 fixed): C[65536,128] = A(bf16) x Bpk(packed bf16), f32 acc.
// Block = 256 threads (8 waves) covering 64 rows x 128 cols.
//  - B pack (32 KB) staged into LDS once per block via the Tensor Data
//    Mover (tensor_load_to_lds + s_wait_tensorcnt), fallback: coop copy.
//  - Each wave keeps its 4 B fragments in registers, reused over 4 M tiles.
//  - Fused epilogue: optionally writes agg = bias + C * dinv(node)^2
//    (GCN self-loop + bias), saving a full 64 MB L2 round-trip per layer.
// =========================================================================
__global__ __launch_bounds__(256)
void gnn_gemm_kernel(const __bf16* __restrict__ A,
                     const __bf16* __restrict__ Bpk,
                     float* __restrict__ C,
                     const float* __restrict__ bias,   // may be null
                     const float* __restrict__ dinv,   // may be null
                     float* __restrict__ agg) {        // may be null
    const int KSTEPS = 4;                               // K = 128
    __shared__ __align__(64) __bf16 sB[8 * KSTEPS * 32 * 16];   // 32 KB

#if GNN_USE_TDM
    if (threadIdx.x < 32) {           // wave 0 issues the TDM transfer
        unsigned long long ga = (unsigned long long)(uintptr_t)Bpk;
        unsigned lds = (unsigned)(uintptr_t)(&sB[0]);   // LDS byte offset
        const unsigned nunits = (unsigned)(sizeof(sB) >> 3);  // 8B units
        u32x4 g0;
        g0[0] = 1u;                                   // count=1 (valid D#)
        g0[1] = lds;                                  // lds_addr
        g0[2] = (unsigned)ga;                         // global_addr[31:0]
        g0[3] = (unsigned)((ga >> 32) & 0x1FFFFFFu)   // global_addr[56:32]
                | 0x80000000u;                        // type=2 ("image")
        i32x8 g1;
        g1[0] = (int)(3u << 16);                      // data_size=8B
        g1[1] = (int)((nunits & 0xFFFFu) << 16);      // tensor_dim0[15:0]
        g1[2] = (int)((nunits >> 16) | (1u << 16));   // td0 hi | tensor_dim1=1
        g1[3] = (int)((nunits & 0xFFFFu) << 16);      // tile_dim0
        g1[4] = 1;                                    // tile_dim1=1
        g1[5] = (int)nunits;                          // tensor_dim0_stride lo
        g1[6] = 0;
        g1[7] = 0;
        i32x4 z4 = {};
#if __has_include(<hip/amd_detail/amd_gfx1250_TDM.h>)
        i32x8 z8 = {};
        __builtin_amdgcn_tensor_load_to_lds(g0, g1, z4, z4, z8, 0);
#else
        __builtin_amdgcn_tensor_load_to_lds(g0, g1, z4, z4, 0);
#endif
        __builtin_amdgcn_s_wait_tensorcnt(0);
    }
#else
    // cooperative copy fallback: 2048 x 16B chunks, 8 per thread
    {
        const float4* src = (const float4*)Bpk;
        float4*       dst = (float4*)sB;
#pragma unroll
        for (int t = 0; t < 8; ++t)
            dst[threadIdx.x + t * 256] = src[threadIdx.x + t * 256];
    }
#endif
    __syncthreads();

    int lane  = threadIdx.x & 31;
    int ntile = threadIdx.x >> 5;          // 0..7
    int n     = ntile * 16 + (lane & 15);

    // B fragments: registers for the whole block's lifetime (4x reuse)
    v16bf bfrag[KSTEPS];
#pragma unroll
    for (int ks = 0; ks < KSTEPS; ++ks)
        bfrag[ks] = *(const v16bf*)(sB + (size_t)((ntile * KSTEPS + ks) * 32 + lane) * 16);

#pragma unroll
    for (int msub = 0; msub < 4; ++msub) {
        int mtile = blockIdx.x * 4 + msub;
        int mrow  = mtile * 16 + (lane & 15);
        v8f acc = {};
#pragma unroll
        for (int ks = 0; ks < KSTEPS; ++ks) {
            int kb0 = ks * 32 + ((lane < 16) ? 0 : 8);
            v8bf alo = *(const v8bf*)(A + (size_t)mrow * CH + kb0);
            v8bf ahi = *(const v8bf*)(A + (size_t)mrow * CH + kb0 + 16);
            v16bf a = __builtin_shufflevector(alo, ahi,
                        0,1,2,3,4,5,6,7,8,9,10,11,12,13,14,15);
            acc = __builtin_amdgcn_wmma_f32_16x16x32_bf16(false, a, false, bfrag[ks],
                                                          (short)0, acc, false, false);
        }
        int mbase = mtile * 16 + ((lane < 16) ? 0 : 8);
        if (agg) {
            float bn = bias[n];
#pragma unroll
            for (int j = 0; j < 8; ++j) {
                int row = mbase + j;
                float v = acc[j];
                C[(size_t)row * CH + n] = v;
                float dn = dinv[row & (N_NODES - 1)];
                agg[(size_t)row * CH + n] = bn + v * dn * dn;
            }
        } else {
#pragma unroll
            for (int j = 0; j < 8; ++j)
                C[(size_t)(mbase + j) * CH + n] = acc[j];
        }
    }
}

// =========================================================================
// Degree / norm over the BASE graph only (graph replicated across batch).
// =========================================================================
__global__ void gnn_deg_init_kernel(float* __restrict__ deg) {
    int nn = blockIdx.x * blockDim.x + threadIdx.x;
    if (nn < N_NODES) deg[nn] = 1.0f;            // self-loop weight
}
__global__ void gnn_deg_accum_kernel(const long long* __restrict__ ei,
                                     const float* __restrict__ ew,
                                     float* __restrict__ deg) {
    int e = blockIdx.x * blockDim.x + threadIdx.x;
    if (e >= E_EDGES) return;
    unsafeAtomicAdd(&deg[(int)ei[E_EDGES + e]], ew[e]);
}
__global__ void gnn_dinv_kernel(float* __restrict__ deg) {
    int nn = blockIdx.x * blockDim.x + threadIdx.x;
    if (nn >= N_NODES) return;
    float d = deg[nn];
    deg[nn] = (d > 0.0f) ? rsqrtf(d) : 0.0f;
}
__global__ void gnn_norm_kernel(const long long* __restrict__ ei,
                                const float* __restrict__ ew,
                                const float* __restrict__ dinv,
                                float* __restrict__ nrm) {
    int e = blockIdx.x * blockDim.x + threadIdx.x;
    if (e >= E_EDGES) return;
    nrm[e] = dinv[(int)ei[e]] * ew[e] * dinv[(int)ei[E_EDGES + e]];
}

// =========================================================================
// Edge scatter: 32 lanes per edge-instance, 4 channels/lane.
// b128 gather from xw[row], 4x global_atomic_add_f32 into agg[col].
// =========================================================================
__global__ __launch_bounds__(256)
void gnn_scatter_kernel(const float* __restrict__ xw,
                        const long long* __restrict__ ei,
                        const float* __restrict__ nrm,
                        float* __restrict__ agg) {
    int tid  = blockIdx.x * blockDim.x + threadIdx.x;
    int inst = tid >> 5;                 // 0 .. B*E-1
    int c0   = (tid & 31) << 2;
    int b    = inst >> 16;               // inst / E_EDGES
    int e    = inst & (E_EDGES - 1);
    int r    = (int)ei[e]            + b * N_NODES;
    int cdst = (int)ei[E_EDGES + e]  + b * N_NODES;
    float w  = nrm[e];
    const float4 v = *(const float4*)(xw + (size_t)r * CH + c0);
    float* dst = agg + (size_t)cdst * CH + c0;
    unsafeAtomicAdd(dst + 0, v.x * w);
    unsafeAtomicAdd(dst + 1, v.y * w);
    unsafeAtomicAdd(dst + 2, v.z * w);
    unsafeAtomicAdd(dst + 3, v.w * w);
}

// =========================================================================
// BN(eval) + ReLU + residual;  writes f32 (residual path) and bf16 (GEMM A).
// =========================================================================
__global__ void gnn_post_kernel(const float* __restrict__ agg,
                                const float* __restrict__ g,  const float* __restrict__ bb,
                                const float* __restrict__ mu, const float* __restrict__ var,
                                float* __restrict__ hf, __bf16* __restrict__ hb) {
    int i = blockIdx.x * blockDim.x + threadIdx.x;
    int c = i & (CH - 1);
    float v = (agg[i] - mu[c]) * rsqrtf(var[c] + BN_EPS) * g[c] + bb[c];
    v = fmaxf(v, 0.0f) + hf[i];          // ReLU + residual (128 -> 128)
    hf[i] = v;
    hb[i] = (__bf16)v;
}

// =========================================================================
// Global mean pool per graph + proj bias:  out[b,c] = mean_n p[b,n,c] + pb[c]
// =========================================================================
__global__ void gnn_reduce_kernel(const float* __restrict__ p,
                                  const float* __restrict__ pb,
                                  float* __restrict__ out) {
    int b = blockIdx.x;          // 16
    int c = threadIdx.x;         // 128
    const float* base = p + (size_t)b * N_NODES * CH + c;
    float s = 0.0f;
    for (int nn = 0; nn < N_NODES; ++nn) {
        __builtin_prefetch(base + (size_t)(nn + 16) * CH, 0, 0);
        s += base[(size_t)nn * CH];
    }
    out[b * CH + c] = s * (1.0f / N_NODES) + pb[c];
}

// =========================================================================
extern "C" void kernel_launch(void* const* d_in, const int* in_sizes, int n_in,
                              void* d_out, int out_size, void* d_ws, size_t ws_size,
                              hipStream_t stream) {
    const float*     x    = (const float*)d_in[0];
    const long long* ei   = (const long long*)d_in[1];   // int64 [2,E]
    const float*     ew   = (const float*)d_in[2];
    const float*     ew1  = (const float*)d_in[3];
    const float*     eb1  = (const float*)d_in[4];
    const float*     ew2  = (const float*)d_in[5];
    const float*     eb2  = (const float*)d_in[6];
    const float*     pw   = (const float*)d_in[7];
    const float*     pb   = (const float*)d_in[8];
    const float*     cw0  = (const float*)d_in[9];
    const float*     cb0  = (const float*)d_in[10];
    const float*     g0   = (const float*)d_in[11];
    const float*     bb0  = (const float*)d_in[12];
    const float*     m0   = (const float*)d_in[13];
    const float*     v0   = (const float*)d_in[14];
    const float*     cw1  = (const float*)d_in[15];
    const float*     cb1  = (const float*)d_in[16];
    const float*     g1   = (const float*)d_in[17];
    const float*     bb1  = (const float*)d_in[18];
    const float*     m1   = (const float*)d_in[19];
    const float*     v1   = (const float*)d_in[20];
    float* out = (float*)d_out;

    // ---- workspace layout (bytes) ----
    char* ws = (char*)d_ws;
    size_t off = 0;
    float*  hf   = (float*) (ws + off); off += (size_t)TOTAL * CH * 4;   // 32 MB
    __bf16* hb   = (__bf16*)(ws + off); off += (size_t)TOTAL * CH * 2;   // 16 MB
    float*  xw   = (float*) (ws + off); off += (size_t)TOTAL * CH * 4;   // 32 MB
    float*  agg  = (float*) (ws + off); off += (size_t)TOTAL * CH * 4;   // 32 MB
    float*  dinv = (float*) (ws + off); off += (size_t)N_NODES * 4;
    float*  nrm  = (float*) (ws + off); off += (size_t)E_EDGES * 4;
    __bf16* w2pk = (__bf16*)(ws + off); off += 8 * 2 * 32 * 16 * 2;      // K=64 pack
    __bf16* w0pk = (__bf16*)(ws + off); off += 8 * 4 * 32 * 16 * 2;      // K=128 pack
    __bf16* w1pk = (__bf16*)(ws + off); off += 8 * 4 * 32 * 16 * 2;
    __bf16* pwpk = (__bf16*)(ws + off); off += 8 * 4 * 32 * 16 * 2;

    const int ELEMS = TOTAL * CH;            // 8,388,608

    // 1) pack weights into WMMA B-fragment layout
    gnn_pack_b_kernel<<<(8*2*32*16 + 255)/256, 256, 0, stream>>>(ew2, w2pk, 64);
    gnn_pack_b_kernel<<<(8*4*32*16 + 255)/256, 256, 0, stream>>>(cw0, w0pk, 128);
    gnn_pack_b_kernel<<<(8*4*32*16 + 255)/256, 256, 0, stream>>>(cw1, w1pk, 128);
    gnn_pack_b_kernel<<<(8*4*32*16 + 255)/256, 256, 0, stream>>>(pw,  pwpk, 128);

    // 2) fused embed (WMMA)
    gnn_embed_kernel<<<TOTAL/16, 256, 0, stream>>>(x, ew1, eb1, w2pk, eb2, hf, hb);

    // 3) GCN normalization on the base graph
    gnn_deg_init_kernel <<<(N_NODES+255)/256, 256, 0, stream>>>(dinv);
    gnn_deg_accum_kernel<<<(E_EDGES+255)/256, 256, 0, stream>>>(ei, ew, dinv);
    gnn_dinv_kernel     <<<(N_NODES+255)/256, 256, 0, stream>>>(dinv);
    gnn_norm_kernel     <<<(E_EDGES+255)/256, 256, 0, stream>>>(ei, ew, dinv, nrm);

    const int gemm_blocks    = TOTAL / 64;                   // 1024
    const int scatter_blocks = (B_GR * E_EDGES * 32) / 256;  // 131072

    // 4) GCN layer 0  (GEMM + fused bias/self-loop epilogue)
    gnn_gemm_kernel    <<<gemm_blocks, 256, 0, stream>>>(hb, w0pk, xw, cb0, dinv, agg);
    gnn_scatter_kernel <<<scatter_blocks, 256, 0, stream>>>(xw, ei, nrm, agg);
    gnn_post_kernel    <<<ELEMS/256, 256, 0, stream>>>(agg, g0, bb0, m0, v0, hf, hb);

    // 5) GCN layer 1
    gnn_gemm_kernel    <<<gemm_blocks, 256, 0, stream>>>(hb, w1pk, xw, cb1, dinv, agg);
    gnn_scatter_kernel <<<scatter_blocks, 256, 0, stream>>>(xw, ei, nrm, agg);
    gnn_post_kernel    <<<ELEMS/256, 256, 0, stream>>>(agg, g1, bb1, m1, v1, hf, hb);

    // 6) projection GEMM + global mean pool
    gnn_gemm_kernel   <<<gemm_blocks, 256, 0, stream>>>(hb, pwpk, xw,
                                                        nullptr, nullptr, nullptr);
    gnn_reduce_kernel <<<B_GR, CH, 0, stream>>>(xw, pb, out);

    (void)in_sizes; (void)n_in; (void)out_size; (void)ws_size;
}
